// Transformer_5995774345320
// MI455X (gfx1250) — matrix operand order repaired and checked by reading, not compiled, not run on hardware
//
#include <hip/hip_runtime.h>

// ---------------------------------------------------------------------------
// Model dims (fixed by the reference)
// ---------------------------------------------------------------------------
#define BB 2
#define SS 1024
#define HH 1024
#define NHH 16
#define FFF 4096
#define LL 4
#define VV 32000
#define HDD 64
#define MT (BB * SS)   // 2048 tokens

typedef __attribute__((ext_vector_type(16))) __bf16 v16bf;
typedef __attribute__((ext_vector_type(8)))  float  v8f;

static __device__ __forceinline__ v8f wmma_bf16(const v16bf& a, const v16bf& b, const v8f& c) {
    return __builtin_amdgcn_wmma_f32_16x16x32_bf16(false, a, false, b, (short)0, c, false, false);
}

// ---------------------------------------------------------------------------
// Embedding + positional encoding (pe indexed by BATCH, per reference)
// ---------------------------------------------------------------------------
__global__ __launch_bounds__(256)
void embed_pe(const int* __restrict__ x, const float* __restrict__ emb,
              float* __restrict__ h) {
    long e = (long)blockIdx.x * 256 + threadIdx.x;      // over B*S*H
    int  hc = (int)(e % HH);
    long ts = e / HH;                                   // token index b*S+s
    int  b  = (int)(ts / SS);
    int  tok = x[ts];
    int   i2    = hc & ~1;
    float div   = __expf(-(float)i2 * (__logf(10000.0f) / (float)HH));
    float ang   = (float)b * div;
    float pe    = (hc & 1) ? __cosf(ang) : __sinf(ang);
    h[e] = emb[(long)tok * HH + hc] + pe;
}

// ---------------------------------------------------------------------------
// GEMM: C[M,N] = A[M,K] @ W[K,N] + bias[N] (+res[M,N]) (+ReLU)
// Block tile 64(M) x 128(N), BK=32, 8 waves; each wave -> 32x32 output
// (2 A-frags x 2 B-frags = 4 WMMA accumulators). LDS double-buffered with
// register-staged global loads so global_load latency overlaps WMMA compute.
// ---------------------------------------------------------------------------
template<bool RELU>
__global__ __launch_bounds__(256)
void gemm_bf16(const float* __restrict__ A, const float* __restrict__ W,
               const float* __restrict__ bias, const float* __restrict__ res,
               float* __restrict__ C, int M, int N, int K) {
    __shared__ __bf16 As[2][64 * 32];    // [buf][row(64) x k(32)]
    __shared__ __bf16 Bs[2][32 * 128];   // [buf][k(32)  x col(128)]

    const int tid  = threadIdx.x;
    const int wv   = tid >> 5;
    const int lane = tid & 31;
    const int wm   = wv & 1;      // M subtile (32 rows)
    const int wn   = wv >> 1;     // N subtile (32 cols)
    const int r    = lane & 15;
    const int hi   = lane >> 4;
    const long mb  = (long)blockIdx.y * 64;
    const long nb  = (long)blockIdx.x * 128;

    float ra[8];     // staged A tile elements (64x32 / 256 threads)
    float rb[16];    // staged B tile elements (32x128 / 256 threads)

    auto load_regs = [&](int k0) {
        #pragma unroll
        for (int i = 0; i < 8; ++i) {
            int idx = tid * 8 + i;
            ra[i] = A[(mb + (idx >> 5)) * (long)K + (k0 + (idx & 31))];
        }
        #pragma unroll
        for (int i = 0; i < 16; ++i) {
            int idx = tid * 16 + i;
            rb[i] = W[(long)(k0 + (idx >> 7)) * N + (nb + (idx & 127))];
        }
    };
    auto store_regs = [&](int buf) {
        #pragma unroll
        for (int i = 0; i < 8; ++i)  As[buf][tid * 8 + i]  = (__bf16)ra[i];
        #pragma unroll
        for (int i = 0; i < 16; ++i) Bs[buf][tid * 16 + i] = (__bf16)rb[i];
    };

    v8f acc00 = {}, acc01 = {}, acc10 = {}, acc11 = {};

    const int nk = K >> 5;
    load_regs(0);
    store_regs(0);
    __syncthreads();

    for (int it = 0; it < nk; ++it) {
        const int cur = it & 1;
        if (it + 1 < nk) load_regs((it + 1) << 5);   // in flight during compute

        v16bf a0, a1, b0, b1;
        #pragma unroll
        for (int e = 0; e < 16; ++e) {               // A 16x32 bf16 layout
            int kk = (e & 7) + ((e >> 3) << 4) + 8 * hi;
            a0[e] = As[cur][(wm * 32 +      r) * 32 + kk];
            a1[e] = As[cur][(wm * 32 + 16 + r) * 32 + kk];
        }
        #pragma unroll
        for (int e = 0; e < 16; ++e) {               // B 32x16 bf16 layout
            int kk = (e & 7) + ((e >> 3) << 4) + 8 * hi;
            b0[e] = Bs[cur][kk * 128 + wn * 32 +      r];
            b1[e] = Bs[cur][kk * 128 + wn * 32 + 16 + r];
        }
        acc00 = wmma_bf16(a0, b0, acc00);
        acc01 = wmma_bf16(a0, b1, acc01);
        acc10 = wmma_bf16(a1, b0, acc10);
        acc11 = wmma_bf16(a1, b1, acc11);

        if (it + 1 < nk) store_regs(cur ^ 1);        // other buffer: no race
        __syncthreads();
    }

    #pragma unroll
    for (int v = 0; v < 8; ++v) {                    // C 16x16 f32 layout
        long r0 = mb + wm * 32 + v + 8 * hi;
        long r1 = r0 + 16;
        long c0 = nb + wn * 32 + r;
        long c1 = c0 + 16;
        float x00 = acc00[v] + bias[c0];
        float x01 = acc01[v] + bias[c1];
        float x10 = acc10[v] + bias[c0];
        float x11 = acc11[v] + bias[c1];
        if (res) {
            x00 += res[r0 * N + c0]; x01 += res[r0 * N + c1];
            x10 += res[r1 * N + c0]; x11 += res[r1 * N + c1];
        }
        if (RELU) {
            x00 = fmaxf(x00, 0.0f); x01 = fmaxf(x01, 0.0f);
            x10 = fmaxf(x10, 0.0f); x11 = fmaxf(x11, 0.0f);
        }
        C[r0 * N + c0] = x00;  C[r0 * N + c1] = x01;
        C[r1 * N + c0] = x10;  C[r1 * N + c1] = x11;
    }
}

// ---------------------------------------------------------------------------
// Flash attention: block = (b, head, 128-query tile); 8 waves x 16 queries.
// K/V staged to LDS as bf16; online softmax; P transposed via LDS to A-layout.
// ---------------------------------------------------------------------------
__global__ __launch_bounds__(256)
void attn(const float* __restrict__ Q, const float* __restrict__ Kg,
          const float* __restrict__ Vg, float* __restrict__ O, float scale) {
    __shared__ __bf16 Ks[32 * 64];
    __shared__ __bf16 Vs[32 * 64];
    __shared__ __bf16 Ps[8 * 16 * 32];   // per-wave 16x32 P tile

    const int tid  = threadIdx.x;
    const int wv   = tid >> 5;
    const int lane = tid & 31;
    const int r    = lane & 15;
    const int hi   = lane >> 4;
    const int b    = blockIdx.z;
    const long hb  = (long)blockIdx.y * HDD;
    const long qrow = (long)blockIdx.x * 128 + wv * 16;
    const long base = (long)b * SS * HH;
    const int  pw   = wv * 512;

    // Q fragments (A 16x32 layout), hd slices [0,32) and [32,64)
    v16bf aq0, aq1;
    #pragma unroll
    for (int e = 0; e < 16; ++e) {
        int kd = (e & 7) + ((e >> 3) << 4) + 8 * hi;
        const float* qp = Q + base + (qrow + r) * HH + hb;
        aq0[e] = (__bf16)qp[kd];
        aq1[e] = (__bf16)qp[32 + kd];
    }

    float m[8], l[8];
    #pragma unroll
    for (int v = 0; v < 8; ++v) { m[v] = -3.0e30f; l[v] = 0.0f; }
    v8f o0 = {}, o1 = {}, o2 = {}, o3 = {};

    for (int kb = 0; kb < SS; kb += 32) {
        __syncthreads();
        #pragma unroll
        for (int i = 0; i < 8; ++i) {               // stage K/V 32x64 tiles
            int idx = tid * 8 + i;
            int kr = idx >> 6, col = idx & 63;
            const float* kp = Kg + base + (long)(kb + kr) * HH + hb;
            const float* vp = Vg + base + (long)(kb + kr) * HH + hb;
            Ks[idx] = (__bf16)kp[col];
            Vs[idx] = (__bf16)vp[col];
        }
        if (kb + 32 < SS) {                          // prefetch next tiles
            int pr = tid >> 3, pc = (tid & 7) * 8;
            __builtin_prefetch(Kg + base + (long)(kb + 32 + pr) * HH + hb + pc, 0, 3);
            __builtin_prefetch(Vg + base + (long)(kb + 32 + pr) * HH + hb + pc, 0, 3);
        }
        __syncthreads();

        // K^T B-fragments: B[kd_slice x 16 keys], element = K[key][kd]
        v16bf bk0, bk1, bk2, bk3;
        #pragma unroll
        for (int e = 0; e < 16; ++e) {
            int kd = (e & 7) + ((e >> 3) << 4) + 8 * hi;
            bk0[e] = Ks[r * 64 + kd];               // keys 0..15, hd 0..31
            bk1[e] = Ks[(16 + r) * 64 + kd];        // keys 16..31, hd 0..31
            bk2[e] = Ks[r * 64 + 32 + kd];          // keys 0..15, hd 32..63
            bk3[e] = Ks[(16 + r) * 64 + 32 + kd];   // keys 16..31, hd 32..63
        }
        v8f s0 = {}, s1 = {};
        s0 = wmma_bf16(aq0, bk0, s0);
        s0 = wmma_bf16(aq1, bk2, s0);
        s1 = wmma_bf16(aq0, bk1, s1);
        s1 = wmma_bf16(aq1, bk3, s1);

        // online softmax per row (rows 0..7 in lanes 0..15, 8..15 in 16..31)
        #pragma unroll
        for (int v = 0; v < 8; ++v) {
            float x0 = s0[v] * scale, x1 = s1[v] * scale;
            float tmax = fmaxf(x0, x1);
            tmax = fmaxf(tmax, __shfl_xor(tmax, 1, 32));
            tmax = fmaxf(tmax, __shfl_xor(tmax, 2, 32));
            tmax = fmaxf(tmax, __shfl_xor(tmax, 4, 32));
            tmax = fmaxf(tmax, __shfl_xor(tmax, 8, 32));
            float mnew = fmaxf(m[v], tmax);
            float p0 = __expf(x0 - mnew), p1 = __expf(x1 - mnew);
            float corr = __expf(m[v] - mnew);
            float rs = p0 + p1;
            rs += __shfl_xor(rs, 1, 32);
            rs += __shfl_xor(rs, 2, 32);
            rs += __shfl_xor(rs, 4, 32);
            rs += __shfl_xor(rs, 8, 32);
            l[v] = l[v] * corr + rs;
            m[v] = mnew;
            o0[v] *= corr; o1[v] *= corr; o2[v] *= corr; o3[v] *= corr;
            Ps[pw + (v + 8 * hi) * 32 + r]      = (__bf16)p0;
            Ps[pw + (v + 8 * hi) * 32 + 16 + r] = (__bf16)p1;
        }
        __syncthreads();

        v16bf ap;                                   // P as A 16x32 fragment
        #pragma unroll
        for (int e = 0; e < 16; ++e) {
            int kk = (e & 7) + ((e >> 3) << 4) + 8 * hi;
            ap[e] = Ps[pw + r * 32 + kk];
        }
        v16bf bv0, bv1, bv2, bv3;                   // V as B 32x16 fragments
        #pragma unroll
        for (int e = 0; e < 16; ++e) {
            int key = (e & 7) + ((e >> 3) << 4) + 8 * hi;
            bv0[e] = Vs[key * 64 + r];
            bv1[e] = Vs[key * 64 + 16 + r];
            bv2[e] = Vs[key * 64 + 32 + r];
            bv3[e] = Vs[key * 64 + 48 + r];
        }
        o0 = wmma_bf16(ap, bv0, o0);
        o1 = wmma_bf16(ap, bv1, o1);
        o2 = wmma_bf16(ap, bv2, o2);
        o3 = wmma_bf16(ap, bv3, o3);
    }

    #pragma unroll
    for (int v = 0; v < 8; ++v) {
        long row = qrow + v + 8 * hi;
        float inv = 1.0f / l[v];
        float* op = O + base + row * HH + hb;
        op[r]      = o0[v] * inv;
        op[16 + r] = o1[v] * inv;
        op[32 + r] = o2[v] * inv;
        op[48 + r] = o3[v] * inv;
    }
}

// ---------------------------------------------------------------------------
// Row LayerNorm over H=1024; one block per row
// ---------------------------------------------------------------------------
__global__ __launch_bounds__(256)
void ln_row(const float* __restrict__ X, const float* __restrict__ g,
            const float* __restrict__ beta, float* __restrict__ Y) {
    __shared__ float red[16];
    long row = blockIdx.x;
    const float* x = X + row * HH;
    float s = 0.0f, s2 = 0.0f;
    for (int i = threadIdx.x; i < HH; i += 256) { float t = x[i]; s += t; s2 += t * t; }
    #pragma unroll
    for (int mask = 16; mask >= 1; mask >>= 1) {
        s  += __shfl_xor(s,  mask, 32);
        s2 += __shfl_xor(s2, mask, 32);
    }
    int wv = threadIdx.x >> 5, lane = threadIdx.x & 31;
    if (lane == 0) { red[wv] = s; red[8 + wv] = s2; }
    __syncthreads();
    if (threadIdx.x == 0) {
        float a = 0.0f, b = 0.0f;
        #pragma unroll
        for (int i = 0; i < 8; ++i) { a += red[i]; b += red[8 + i]; }
        red[0] = a; red[8] = b;
    }
    __syncthreads();
    float mean = red[0] * (1.0f / HH);
    float var  = red[8] * (1.0f / HH) - mean * mean;
    float rstd = rsqrtf(var + 1e-5f);
    for (int i = threadIdx.x; i < HH; i += 256)
        Y[row * HH + i] = (x[i] - mean) * rstd * g[i] + beta[i];
}

// ---------------------------------------------------------------------------
// Launcher
// ---------------------------------------------------------------------------
extern "C" void kernel_launch(void* const* d_in, const int* in_sizes, int n_in,
                              void* d_out, int out_size, void* d_ws, size_t ws_size,
                              hipStream_t stream) {
    const int*   x     = (const int*)  d_in[0];
    const float* emb   = (const float*)d_in[1];
    const float* Wq    = (const float*)d_in[2];
    const float* bq    = (const float*)d_in[3];
    const float* Wk    = (const float*)d_in[4];
    const float* bk    = (const float*)d_in[5];
    const float* Wv    = (const float*)d_in[6];
    const float* bv    = (const float*)d_in[7];
    const float* Wo    = (const float*)d_in[8];
    const float* bo    = (const float*)d_in[9];
    const float* ln1g  = (const float*)d_in[10];
    const float* ln1b  = (const float*)d_in[11];
    const float* W1    = (const float*)d_in[12];
    const float* b1    = (const float*)d_in[13];
    const float* W2    = (const float*)d_in[14];
    const float* b2    = (const float*)d_in[15];
    const float* ln2g  = (const float*)d_in[16];
    const float* ln2b  = (const float*)d_in[17];
    const float* fcw   = (const float*)d_in[18];
    const float* fcb   = (const float*)d_in[19];
    float* out = (float*)d_out;

    float* ws   = (float*)d_ws;
    const long TOK = (long)MT * HH;    // 2M floats
    float* h    = ws;
    float* tmp  = ws + TOK;
    float* q    = ws + TOK * 2;
    float* k    = ws + TOK * 3;
    float* v    = ws + TOK * 4;
    float* ao   = ws + TOK * 5;
    float* ff   = ws + TOK * 6;        // 2048 x 4096

    const float scale = 1.0f / 8.0f;   // 1/sqrt(64)

    embed_pe<<<(MT * HH) / 256, 256, 0, stream>>>(x, emb, h);

    dim3 blk(256);
    dim3 gHH(HH / 128, MT / 64);       // N=1024 : 8 x 32
    dim3 gFF(FFF / 128, MT / 64);      // N=4096 : 32 x 32
    dim3 gVV(VV / 128, MT / 64);       // N=32000: 250 x 32
    dim3 gAttn(SS / 128, NHH, BB);

    for (int li = 0; li < LL; ++li) {
        const float* wq = Wq + (long)li * HH * HH;
        const float* wk = Wk + (long)li * HH * HH;
        const float* wv_ = Wv + (long)li * HH * HH;
        const float* wo = Wo + (long)li * HH * HH;
        const float* w1 = W1 + (long)li * HH * FFF;
        const float* w2 = W2 + (long)li * FFF * HH;

        gemm_bf16<false><<<gHH, blk, 0, stream>>>(h, wq, bq + li * HH, nullptr, q, MT, HH, HH);
        gemm_bf16<false><<<gHH, blk, 0, stream>>>(h, wk, bk + li * HH, nullptr, k, MT, HH, HH);
        gemm_bf16<false><<<gHH, blk, 0, stream>>>(h, wv_, bv + li * HH, nullptr, v, MT, HH, HH);

        attn<<<gAttn, blk, 0, stream>>>(q, k, v, ao, scale);

        gemm_bf16<false><<<gHH, blk, 0, stream>>>(ao, wo, bo + li * HH, h, tmp, MT, HH, HH);
        ln_row<<<MT, blk, 0, stream>>>(tmp, ln1g + li * HH, ln1b + li * HH, h);

        gemm_bf16<true ><<<gFF, blk, 0, stream>>>(h, w1, b1 + li * FFF, nullptr, ff, MT, FFF, HH);
        gemm_bf16<false><<<gHH, blk, 0, stream>>>(ff, w2, b2 + li * HH, h, tmp, MT, HH, FFF);
        ln_row<<<MT, blk, 0, stream>>>(tmp, ln2g + li * HH, ln2b + li * HH, h);
    }

    gemm_bf16<false><<<gVV, blk, 0, stream>>>(h, fcw, fcb, nullptr, out, MT, VV, HH);
}
